// Attention_14130442404150
// MI455X (gfx1250) — matrix operand order repaired and checked by reading, not compiled, not run on hardware
//
#include <hip/hip_runtime.h>
#include <hip/hip_bf16.h>
#include <math.h>

typedef __attribute__((ext_vector_type(16))) _Float16 v16h;
typedef __attribute__((ext_vector_type(8)))  _Float16 v8h;
typedef __attribute__((ext_vector_type(8)))  float    v8f;
typedef __attribute__((ext_vector_type(4)))  unsigned int v4u;
typedef __attribute__((ext_vector_type(8)))  int      v8i;
typedef __attribute__((ext_vector_type(4)))  int      v4i;

#define B_  2
#define S_  2048
#define D_  1024
#define H_  16
#define DH_ 64
#define NEGBIG (-1.0e30f)

#if __has_builtin(__builtin_amdgcn_tensor_load_to_lds)
#define HAVE_TDM 1
#else
#define HAVE_TDM 0
#endif

union F8  { v8f v; float f[8]; };
union H16 { v16h v; v8h h8[2]; _Float16 h[16]; };

// 16 contiguous halves (B-fragment: lane n holds K = hi*16 + e contiguously).
__device__ __forceinline__ v16h load16(const _Float16* p) {
  H16 r;
  r.h8[0] = *(const v8h*)(p);
  r.h8[1] = *(const v8h*)(p + 8);
  return r.v;
}

// A-fragment (16x32 f16): lane<16: K{0..15}; lane>=16: K{16..31}.
__device__ __forceinline__ v16h load_a(const _Float16* row, int hi) {
  H16 r;
  r.h8[0] = *(const v8h*)(row + hi * 16);
  r.h8[1] = *(const v8h*)(row + hi * 16 + 8);
  return r.v;
}

__device__ __forceinline__ v8f wmma32(v16h a, v16h b, v8f c) {
  return __builtin_amdgcn_wmma_f32_16x16x32_f16(false, a, false, b, (short)0, c,
                                                false, false);
}

__device__ __forceinline__ float rowmax16(float v) {
#pragma unroll
  for (int m = 1; m < 16; m <<= 1) v = fmaxf(v, __shfl_xor(v, m, 32));
  return v;
}
__device__ __forceinline__ float rowsum16(float v) {
#pragma unroll
  for (int m = 1; m < 16; m <<= 1) v += __shfl_xor(v, m, 32);
  return v;
}

#if HAVE_TDM
// 2D Tensor-Data-Mover load: tile (tile_d1 rows x tile_d0 elems) of 2-byte data
// from row-major global (row stride stride0 elems) into packed LDS at lds_off.
// D# per CDNA5 ISA 8.3-8.5: group0 = {count/type, lds, gaddr}, group1 = dims.
// clang-23 toolchain: 6-arg form (g0, g1, g2, g3, g4, cpol).
__device__ __forceinline__ void tdm_load_2d(unsigned lds_off, const void* gaddr,
                                            unsigned tile_d0, unsigned tile_d1,
                                            unsigned tensor_d0, unsigned tensor_d1,
                                            unsigned long long stride0) {
  unsigned long long ga = (unsigned long long)(uintptr_t)gaddr;
  v4u g0;
  g0[0] = 1u;                                           // count=1, user D#
  g0[1] = lds_off;                                      // lds_addr (bytes)
  g0[2] = (unsigned)(ga & 0xffffffffu);                 // global_addr[31:0]
  g0[3] = (unsigned)((ga >> 32) & 0x1ffffffu) | (2u << 30); // addr[56:32] | type=2
  v8i g1;
  g1[0] = (int)(1u << 16);                              // data_size=1 (2 bytes)
  g1[1] = (int)((tensor_d0 & 0xffffu) << 16);           // tensor_dim0[15:0]
  g1[2] = (int)((tensor_d0 >> 16) | ((tensor_d1 & 0xffffu) << 16));
  g1[3] = (int)((tensor_d1 >> 16) | (tile_d0 << 16));   // tile_dim0
  g1[4] = (int)(tile_d1);                               // tile_dim1, tile_dim2=0
  g1[5] = (int)(stride0 & 0xffffffffu);                 // dim0_stride[31:0]
  g1[6] = (int)((stride0 >> 32) & 0xffffu);             // dim0_stride[47:32]
  g1[7] = 0;
  v4i z4 = {0, 0, 0, 0};
  v8i z8 = {0, 0, 0, 0, 0, 0, 0, 0};
  __builtin_amdgcn_tensor_load_to_lds(g0, g1, z4, z4, z8, 0);
}
#endif

// ---------------- f32 -> f16 conversion ----------------
__global__ void k_cvt(const float* __restrict__ src, _Float16* __restrict__ dst, int n) {
  int i = blockIdx.x * blockDim.x + threadIdx.x;
  int stride = gridDim.x * blockDim.x;
  for (; i < n; i += stride) dst[i] = (_Float16)src[i];
}

// ---------------- QKV projection + RoPE ----------------
// qkv[m,n] = sum_k x[m,k]*wQKV[n,k]; 64x64 per wave (4x4 register blocking).
__global__ __launch_bounds__(256) void k_qkv(const _Float16* __restrict__ xh,
                                             const _Float16* __restrict__ wh,
                                             const float* __restrict__ cosp,
                                             const float* __restrict__ sinp,
                                             _Float16* __restrict__ Qh,
                                             _Float16* __restrict__ Kh,
                                             _Float16* __restrict__ Vt) {
  int wave = threadIdx.x >> 5, lane = threadIdx.x & 31;
  int nl = lane & 15, hi = lane >> 4;
  int nt = blockIdx.x;                 // 0..47 : 64-wide N block
  int mt = blockIdx.y * 8 + wave;      // 0..63 : 64-tall M block

  const _Float16* arow0 = xh + (size_t)(mt * 64 + nl) * D_;
  const _Float16* brow0 = wh + (size_t)(nt * 64 + nl) * D_;

  F8 acc[4][4];
#pragma unroll
  for (int mi = 0; mi < 4; ++mi)
#pragma unroll
    for (int ni = 0; ni < 4; ++ni)
#pragma unroll
      for (int r = 0; r < 8; ++r) acc[mi][ni].f[r] = 0.0f;

  for (int kc = 0; kc < D_; kc += 32) {
    v16h a[4], b[4];
#pragma unroll
    for (int mi = 0; mi < 4; ++mi) a[mi] = load_a(arow0 + (size_t)mi * 16 * D_ + kc, hi);
#pragma unroll
    for (int ni = 0; ni < 4; ++ni) b[ni] = load16(brow0 + (size_t)ni * 16 * D_ + kc + hi * 16);
#pragma unroll
    for (int mi = 0; mi < 4; ++mi)
#pragma unroll
      for (int ni = 0; ni < 4; ++ni)
        acc[mi][ni].v = wmma32(a[mi], b[ni], acc[mi][ni].v);
  }

#pragma unroll
  for (int ni = 0; ni < 4; ++ni) {
    int ng = nt * 64 + ni * 16 + nl;   // global column in [0,3072)
    int region = ng >> 10;             // 0=q 1=k 2=v (uniform across 16 lanes)
    int cin = ng & 1023;
    int h = cin >> 6;
    int d = cin & 63;
#pragma unroll
    for (int mi = 0; mi < 4; ++mi) {
#pragma unroll
      for (int r = 0; r < 8; ++r) {
        int gm = mt * 64 + mi * 16 + r + 8 * hi;
        int b = gm >> 11;
        int s = gm & (S_ - 1);
        float val = acc[mi][ni].f[r];
        if (region < 2) {
          float partner = __shfl_xor(val, 1, 32);   // adjacent column = rope pair
          int i = d >> 1;
          float cs = cosp[s * 32 + i];
          float sn = sinp[s * 32 + i];
          val = (d & 1) ? (val * cs + partner * sn) : (val * cs - partner * sn);
          size_t idx = ((size_t)(b * H_ + h) * S_ + s) * DH_ + d;
          if (region == 0) Qh[idx] = (_Float16)val;
          else             Kh[idx] = (_Float16)val;
        } else {
          Vt[((size_t)(b * H_ + h) * DH_ + d) * S_ + s] = (_Float16)val;
        }
      }
    }
  }
}

// ---------------- Causal flash attention ----------------
// One wave per 16-row Q block; kv tiles of 32; K/V staged in LDS by the TDM.
__global__ __launch_bounds__(128) void k_attn(const _Float16* __restrict__ Qh,
                                              const _Float16* __restrict__ Kh,
                                              const _Float16* __restrict__ Vt,
                                              _Float16* __restrict__ attnh) {
  // per-wave: K tile 32x64 (2048h) + V tile 64x32 (2048h) + P tile 16x32 (512h)
  __shared__ __align__(16) _Float16 sh[4 * 4608];
  int wave = threadIdx.x >> 5, lane = threadIdx.x & 31;
  int nl = lane & 15, hi = lane >> 4;
  int b = blockIdx.z, h = blockIdx.y;
  int qt = blockIdx.x * 4 + wave;
  int qs = qt * 16;

  const _Float16* Qb = Qh + (size_t)(b * H_ + h) * S_ * DH_;
  const _Float16* Kb = Kh + (size_t)(b * H_ + h) * S_ * DH_;
  const _Float16* Vb = Vt + (size_t)(b * H_ + h) * DH_ * S_;
  _Float16* myK = sh + wave * 4608;
  _Float16* myV = myK + 2048;
  _Float16* myP = myV + 2048;

  v16h aQ0 = load_a(Qb + (size_t)(qs + nl) * DH_ + 0, hi);
  v16h aQ1 = load_a(Qb + (size_t)(qs + nl) * DH_ + 32, hi);

  F8 o[4];
#pragma unroll
  for (int n = 0; n < 4; ++n)
#pragma unroll
    for (int r = 0; r < 8; ++r) o[n].f[r] = 0.0f;

  float mrow[8], lrow[8];
#pragma unroll
  for (int r = 0; r < 8; ++r) { mrow[r] = NEGBIG; lrow[r] = 0.0f; }

  int nkv = (qs + 16 + 31) >> 5;
  for (int t = 0; t < nkv; ++t) {
    int kv = t * 32;

#if HAVE_TDM
    // Stage K (32 rows x 64) and V^T (64 rows x 32) tiles into LDS via the TDM.
    tdm_load_2d((unsigned)(uintptr_t)myK, Kb + (size_t)kv * DH_,
                /*tile*/ DH_, 32, /*tensor*/ DH_, S_, /*stride0*/ DH_);
    tdm_load_2d((unsigned)(uintptr_t)myV, Vb + kv,
                /*tile*/ 32, DH_, /*tensor*/ S_, DH_, /*stride0*/ S_);
    __builtin_amdgcn_s_wait_tensorcnt(0);
#else
    // Fallback: cooperative vector copy into the same LDS layout.
#pragma unroll
    for (int i = 0; i < 8; ++i) {
      int row = i * 4 + (lane >> 3);
      int col = (lane & 7) * 8;
      *(v8h*)(myK + row * DH_ + col) = *(const v8h*)(Kb + (size_t)(kv + row) * DH_ + col);
    }
#pragma unroll
    for (int i = 0; i < 8; ++i) {
      int row = i * 8 + (lane >> 2);
      int col = (lane & 3) * 8;
      *(v8h*)(myV + row * 32 + col) = *(const v8h*)(Vb + (size_t)row * S_ + kv + col);
    }
    asm volatile("s_wait_dscnt 0" ::: "memory");
#endif

    // S tile 16x32 = Q @ K^T from LDS
    F8 sc[2];
#pragma unroll
    for (int jh = 0; jh < 2; ++jh) {
      F8 s;
#pragma unroll
      for (int r = 0; r < 8; ++r) s.f[r] = 0.0f;
      const _Float16* krow = myK + (jh * 16 + nl) * DH_;
      s.v = wmma32(aQ0, load16(krow + 0 + hi * 16), s.v);
      s.v = wmma32(aQ1, load16(krow + 32 + hi * 16), s.v);
      sc[jh] = s;
    }

    // Online softmax (rows r + 8*hi live on this half-wave's 16 lanes)
    float corr[8];
#pragma unroll
    for (int r = 0; r < 8; ++r) {
      int srow = qs + r + 8 * hi;
#pragma unroll
      for (int jh = 0; jh < 2; ++jh) {
        int j = kv + jh * 16 + nl;
        float v = sc[jh].f[r] * 0.125f;      // 1/sqrt(64)
        if (j > srow) v = NEGBIG;            // causal mask
        sc[jh].f[r] = v;
      }
      float vmax = rowmax16(fmaxf(sc[0].f[r], sc[1].f[r]));
      float mn = fmaxf(mrow[r], vmax);
      corr[r] = __expf(mrow[r] - mn);
      float p0 = __expf(sc[0].f[r] - mn);
      float p1 = __expf(sc[1].f[r] - mn);
      sc[0].f[r] = p0; sc[1].f[r] = p1;
      lrow[r] = lrow[r] * corr[r] + rowsum16(p0 + p1);
      mrow[r] = mn;
    }
#pragma unroll
    for (int n = 0; n < 4; ++n)
#pragma unroll
      for (int r = 0; r < 8; ++r) o[n].f[r] *= corr[r];

    // Transpose P (C-layout) -> A-layout through per-wave LDS slice
#pragma unroll
    for (int jh = 0; jh < 2; ++jh)
#pragma unroll
      for (int r = 0; r < 8; ++r)
        myP[(r + 8 * hi) * 32 + jh * 16 + nl] = (_Float16)sc[jh].f[r];
    asm volatile("s_wait_dscnt 0" ::: "memory");
    v16h aP = load_a(myP + nl * 32, hi);

    // O += P @ V from LDS (V stored transposed: contiguous kv per dh row)
#pragma unroll
    for (int n = 0; n < 4; ++n) {
      o[n].v = wmma32(aP, load16(myV + (n * 16 + nl) * 32 + hi * 16), o[n].v);
    }
  }

  // Normalize and write attn output as (B,S,D) f16 for the output projection
#pragma unroll
  for (int r = 0; r < 8; ++r) {
    float inv = 1.0f / lrow[r];
    size_t rowbase = ((size_t)b * S_ + (qs + r + 8 * hi)) * D_ + h * DH_;
#pragma unroll
    for (int n = 0; n < 4; ++n)
      attnh[rowbase + n * 16 + nl] = (_Float16)(o[n].f[r] * inv);
  }
}

// ---------------- Output projection ----------------
// out[m,n] = sum_k attn[m,k]*wOut[n,k]; 64x64 per wave (4x4 blocking) -> f32.
__global__ __launch_bounds__(256) void k_out(const _Float16* __restrict__ ah,
                                             const _Float16* __restrict__ wh,
                                             float* __restrict__ out) {
  int wave = threadIdx.x >> 5, lane = threadIdx.x & 31;
  int nl = lane & 15, hi = lane >> 4;
  int nt = blockIdx.x;                 // 0..15
  int mt = blockIdx.y * 8 + wave;      // 0..63

  const _Float16* arow0 = ah + (size_t)(mt * 64 + nl) * D_;
  const _Float16* brow0 = wh + (size_t)(nt * 64 + nl) * D_;

  F8 acc[4][4];
#pragma unroll
  for (int mi = 0; mi < 4; ++mi)
#pragma unroll
    for (int ni = 0; ni < 4; ++ni)
#pragma unroll
      for (int r = 0; r < 8; ++r) acc[mi][ni].f[r] = 0.0f;

  for (int kc = 0; kc < D_; kc += 32) {
    v16h a[4], b[4];
#pragma unroll
    for (int mi = 0; mi < 4; ++mi) a[mi] = load_a(arow0 + (size_t)mi * 16 * D_ + kc, hi);
#pragma unroll
    for (int ni = 0; ni < 4; ++ni) b[ni] = load16(brow0 + (size_t)ni * 16 * D_ + kc + hi * 16);
#pragma unroll
    for (int mi = 0; mi < 4; ++mi)
#pragma unroll
      for (int ni = 0; ni < 4; ++ni)
        acc[mi][ni].v = wmma32(a[mi], b[ni], acc[mi][ni].v);
  }

#pragma unroll
  for (int mi = 0; mi < 4; ++mi)
#pragma unroll
    for (int ni = 0; ni < 4; ++ni)
#pragma unroll
      for (int r = 0; r < 8; ++r)
        out[(size_t)(mt * 64 + mi * 16 + r + 8 * hi) * D_ + nt * 64 + ni * 16 + nl] =
            acc[mi][ni].f[r];
}

extern "C" void kernel_launch(void* const* d_in, const int* in_sizes, int n_in,
                              void* d_out, int out_size, void* d_ws, size_t ws_size,
                              hipStream_t stream) {
  (void)in_sizes; (void)n_in; (void)out_size; (void)ws_size;
  const float* x    = (const float*)d_in[0];
  const float* wQKV = (const float*)d_in[1];
  const float* wOut = (const float*)d_in[2];
  const float* cosp = (const float*)d_in[3];
  const float* sinp = (const float*)d_in[4];
  float* out = (float*)d_out;

  const size_t XN  = (size_t)B_ * S_ * D_;   // 4,194,304
  const size_t WQN = (size_t)3 * D_ * D_;    // 3,145,728
  const size_t WON = (size_t)D_ * D_;        // 1,048,576

  _Float16* wsp = (_Float16*)d_ws;
  _Float16* xh  = wsp;
  _Float16* wqh = xh + XN;
  _Float16* woh = wqh + WQN;
  _Float16* Qh  = woh + WON;
  _Float16* Kh  = Qh + XN;
  _Float16* Vt  = Kh + XN;
  _Float16* Ah  = Vt + XN;

  k_cvt<<<1024, 256, 0, stream>>>(x, xh, (int)XN);
  k_cvt<<<1024, 256, 0, stream>>>(wQKV, wqh, (int)WQN);
  k_cvt<<<512, 256, 0, stream>>>(wOut, woh, (int)WON);

  // QKV GEMM + RoPE: N blocks = 3072/64 = 48, M blocks = 4096/64 = 64 (8/block)
  k_qkv<<<dim3(48, 8, 1), 256, 0, stream>>>(xh, wqh, cosp, sinp, Qh, Kh, Vt);

  // Flash attention: 128 q-tiles per (b,h), 4 waves per block
  k_attn<<<dim3(32, 16, 2), 128, 0, stream>>>(Qh, Kh, Vt, Ah);

  // Output projection: N blocks = 16, M blocks = 64 (8 per block)
  k_out<<<dim3(16, 8, 1), 256, 0, stream>>>(Ah, woh, out);
}
